// DSGIAT_GraphBranch_25142738551481
// MI455X (gfx1250) — compile-verified, matrix operand order checked by reading, not run on hardware
//
#include <hip/hip_runtime.h>
#include <hip/hip_bf16.h>

// ---------------------------------------------------------------------------
// Problem constants (match reference)
// ---------------------------------------------------------------------------
#define NN      50000
#define EE      800000
#define IN_CH   128
#define HID     64
#define HEADS   4
#define GG      64
#define HD      (HEADS * HID)       // 256
#define DCAT    (IN_CH + 2 * HD)    // 640
#define NEG_SLOPE 0.2f

typedef __attribute__((ext_vector_type(16))) __bf16 v16bf;
typedef __attribute__((ext_vector_type(8)))  float  v8f;

#if defined(__AMDGCN__) && __has_builtin(__builtin_amdgcn_global_load_async_to_lds_b128) && __has_builtin(__builtin_amdgcn_s_wait_asynccnt)
#define GEMM_ASYNC 1
// exact parameter types per hipcc diagnostic: vector_size(16) int, AS1 src / AS3 dst
typedef int v4i_gcc __attribute__((vector_size(16)));
typedef __attribute__((address_space(1))) v4i_gcc* as1_v4i;
typedef __attribute__((address_space(3))) v4i_gcc* as3_v4i;
#else
#define GEMM_ASYNC 0
#endif

static __device__ __forceinline__ unsigned short f2bf(float f) {
    unsigned int u = __float_as_uint(f);
    unsigned int r = (u + 0x7FFFu + ((u >> 16) & 1u)) >> 16;  // RNE
    return (unsigned short)r;
}

// ---------------------------------------------------------------------------
// Elementwise helpers
// ---------------------------------------------------------------------------
__global__ void k_f32_to_bf16(const float* __restrict__ in,
                              unsigned short* __restrict__ out, int n) {
    int i = blockIdx.x * blockDim.x + threadIdx.x;
    if (i < n) out[i] = f2bf(in[i]);
}

// Wt[n*K + k] = bf16(W[k*N + n])   (weights are small; done once per layer)
__global__ void k_transpose_bf16(const float* __restrict__ W,
                                 unsigned short* __restrict__ Wt, int K, int N) {
    int i = blockIdx.x * blockDim.x + threadIdx.x;
    if (i >= K * N) return;
    int n = i / K, k = i - n * K;
    Wt[i] = f2bf(W[(size_t)k * N + n]);
}

__global__ void k_fill_f32(float* __restrict__ p, float v, int n) {
    int i = blockIdx.x * blockDim.x + threadIdx.x;
    if (i < n) p[i] = v;
}

// ---------------------------------------------------------------------------
// WMMA bf16 GEMM: C[M,N] = A[M,K] * Bt[N,K]^T
//   A  : [M,K] bf16 row-major,  Bt : [N,K] bf16 row-major (pre-transposed B)
// block = 256 threads (8 waves), 64x128 tile; wave (wr=w>>2, wc=w&3) computes
// a 32x32 sub-tile = 2x2 WMMA fragments (4 accumulators, 4 WMMA / K-step).
// Staging is pure 16B block copies (async-to-LDS when available).
// Requires: K % 32 == 0, N % 128 == 0 (M may be ragged, M >= 1).
// ---------------------------------------------------------------------------
__global__ void k_gemm_bf16_wmma(const unsigned short* __restrict__ A,
                                 const unsigned short* __restrict__ Bt,
                                 float* __restrict__ C,
                                 unsigned short* __restrict__ Cbf,
                                 const float* __restrict__ bias,
                                 int M, int N, int K, int relu) {
    __shared__ __align__(16) unsigned short As[64 * 40];   // [m_local][k]
    __shared__ __align__(16) unsigned short Bs[128 * 40];  // [n_local][k]

    const int t    = threadIdx.x;
    const int wave = t >> 5;
    const int lane = t & 31;
    const int wr   = wave >> 2;          // 0..1
    const int wc   = wave & 3;           // 0..3
    const int m0   = blockIdx.x * 64;
    const int n0   = blockIdx.y * 128;

    v8f c00 = {0.f,0.f,0.f,0.f,0.f,0.f,0.f,0.f};
    v8f c01 = c00, c10 = c00, c11 = c00;

    const int rr    = lane & 15;
    const int kbase = (lane >> 4) * 8;   // halfs

    // staging coordinates
    const int ar  = t >> 2;              // 0..63  (A row)
    const int ach = (t & 3) * 8;         // 0,8,16,24 halfs
    const int agm = min(m0 + ar, M - 1); // clamp: padded rows feed only padded C rows

    for (int kb = 0; kb < K; kb += 32) {
        // ---- stage A tile 64x32 + B tile 128x32 (16B chunks) ----
#if GEMM_ASYNC
        __builtin_amdgcn_global_load_async_to_lds_b128(
            (as1_v4i)(A + (size_t)agm * K + kb + ach),
            (as3_v4i)(&As[ar * 40 + ach]), 0, 0);
        #pragma unroll
        for (int i = 0; i < 2; ++i) {
            int idx = t + 256 * i;
            int br  = idx >> 2;          // 0..127
            int bch = (idx & 3) * 8;
            __builtin_amdgcn_global_load_async_to_lds_b128(
                (as1_v4i)(Bt + (size_t)(n0 + br) * K + kb + bch),
                (as3_v4i)(&Bs[br * 40 + bch]), 0, 0);
        }
        __builtin_amdgcn_s_wait_asynccnt(0);
#else
        *(uint4*)(&As[ar * 40 + ach]) =
            *(const uint4*)(A + (size_t)agm * K + kb + ach);
        #pragma unroll
        for (int i = 0; i < 2; ++i) {
            int idx = t + 256 * i;
            int br  = idx >> 2;
            int bch = (idx & 3) * 8;
            *(uint4*)(&Bs[br * 40 + bch]) =
                *(const uint4*)(Bt + (size_t)(n0 + br) * K + kb + bch);
        }
#endif
        __syncthreads();

        // ---- build fragments (two ds_load_b128 each) and issue 4 WMMAs ----
        union Frag { uint4 u[2]; v16bf v; } a0, a1, b0, b1;
        const unsigned short* p;
        p = &As[(wr * 32 +      rr) * 40]; a0.u[0] = *(const uint4*)(p + kbase); a0.u[1] = *(const uint4*)(p + kbase + 16);
        p = &As[(wr * 32 + 16 + rr) * 40]; a1.u[0] = *(const uint4*)(p + kbase); a1.u[1] = *(const uint4*)(p + kbase + 16);
        p = &Bs[(wc * 32 +      rr) * 40]; b0.u[0] = *(const uint4*)(p + kbase); b0.u[1] = *(const uint4*)(p + kbase + 16);
        p = &Bs[(wc * 32 + 16 + rr) * 40]; b1.u[0] = *(const uint4*)(p + kbase); b1.u[1] = *(const uint4*)(p + kbase + 16);

        c00 = __builtin_amdgcn_wmma_f32_16x16x32_bf16(false, a0.v, false, b0.v, (short)0, c00, false, false);
        c01 = __builtin_amdgcn_wmma_f32_16x16x32_bf16(false, a0.v, false, b1.v, (short)0, c01, false, false);
        c10 = __builtin_amdgcn_wmma_f32_16x16x32_bf16(false, a1.v, false, b0.v, (short)0, c10, false, false);
        c11 = __builtin_amdgcn_wmma_f32_16x16x32_bf16(false, a1.v, false, b1.v, (short)0, c11, false, false);
        __syncthreads();
    }

    // ---- epilogue: C layout VGPR r, lane l -> row (l>>4)*8 + r, col l&15 ----
    const v8f* accs[4] = { &c00, &c01, &c10, &c11 };
    #pragma unroll
    for (int sm = 0; sm < 2; ++sm) {
        #pragma unroll
        for (int sn = 0; sn < 2; ++sn) {
            const v8f& acc = *accs[sm * 2 + sn];
            const int col  = n0 + wc * 32 + sn * 16 + (lane & 15);
            const int rowb = m0 + wr * 32 + sm * 16 + (lane >> 4) * 8;
            const float bv = bias ? bias[col] : 0.f;
            #pragma unroll
            for (int r = 0; r < 8; ++r) {
                int gm = rowb + r;
                if (gm < M) {
                    float v = acc[r] + bv;
                    if (relu) v = fmaxf(v, 0.f);
                    C[(size_t)gm * N + col] = v;
                    if (Cbf) Cbf[(size_t)gm * N + col] = f2bf(v);
                }
            }
        }
    }
}

// ---------------------------------------------------------------------------
// Attention scaffolding
// ---------------------------------------------------------------------------
__global__ void k_node_scores(const float* __restrict__ h,   // [N, 256]
                              const float* __restrict__ asrc,// [4*64]
                              const float* __restrict__ adst,// [4*64]
                              float* __restrict__ es,        // [N,4]
                              float* __restrict__ ed, int n) {
    int i = blockIdx.x * blockDim.x + threadIdx.x;
    if (i >= n * HEADS) return;
    int node = i >> 2, hh = i & 3;
    const float* hp = h + (size_t)node * HD + hh * HID;
    const float* as = asrc + hh * HID;
    const float* ad = adst + hh * HID;
    float ss = 0.f, sd = 0.f;
    #pragma unroll 8
    for (int d = 0; d < HID; ++d) { float v = hp[d]; ss += v * as[d]; sd += v * ad[d]; }
    es[i] = ss;
    ed[i] = sd;
}

static __device__ __forceinline__ void atomicMaxF(float* addr, float val) {
    if (val >= 0.f) atomicMax((int*)addr, __float_as_int(val));
    else            atomicMin((unsigned int*)addr, __float_as_uint(val));
}

__global__ void k_edge_lrelu_max(const int* __restrict__ ei,   // [2,E]
                                 const float* __restrict__ es,
                                 const float* __restrict__ ed,
                                 float* __restrict__ ebuf,     // [E,4]
                                 float* __restrict__ m,        // [N,4] init -inf
                                 int e_total) {
    int i = blockIdx.x * blockDim.x + threadIdx.x;
    if (i >= e_total) return;
    int e = i >> 2, hh = i & 3;
    int s = ei[e], d = ei[EE + e];
    float v = es[s * HEADS + hh] + ed[d * HEADS + hh];
    v = (v > 0.f) ? v : NEG_SLOPE * v;
    ebuf[i] = v;
    atomicMaxF(&m[d * HEADS + hh], v);
}

__global__ void k_fix_m(float* __restrict__ m, int n) {
    int i = blockIdx.x * blockDim.x + threadIdx.x;
    if (i >= n) return;
    float v = m[i];
    if (!isfinite(v)) v = 0.f;
    m[i] = v;
}

__global__ void k_edge_exp_sum(const int* __restrict__ ei,
                               const float* __restrict__ m,
                               float* __restrict__ ebuf,
                               float* __restrict__ denom,     // [N,4] init 0
                               int e_total) {
    int i = blockIdx.x * blockDim.x + threadIdx.x;
    if (i >= e_total) return;
    int e = i >> 2, hh = i & 3;
    int d = ei[EE + e];
    float ex = __expf(ebuf[i] - m[d * HEADS + hh]);
    ebuf[i] = ex;
    atomicAdd(&denom[d * HEADS + hh], ex);
}

// one block per edge: src/dst loads are block-uniform -> scalar loads
__global__ void k_edge_scatter(const int* __restrict__ ei,
                               const float* __restrict__ hpre, // [N,256]
                               const float* __restrict__ ebuf, // [E,4] = exp
                               const float* __restrict__ denom,// [N,4]
                               float* __restrict__ agg) {      // [N,256] init 0
    int e  = blockIdx.x;         // 0..E-1
    int c  = threadIdx.x;        // 0..255
    int hh = c >> 6;
    int s = ei[e], d = ei[EE + e];
    float alpha = ebuf[e * HEADS + hh] / (denom[d * HEADS + hh] + 1e-16f);
    atomicAdd(&agg[(size_t)d * HD + c], hpre[(size_t)s * HD + c] * alpha);
}

__global__ void k_bias_relu(const float* __restrict__ agg,
                            const float* __restrict__ bias,
                            float* __restrict__ out,
                            unsigned short* __restrict__ outbf, int total) {
    int i = blockIdx.x * blockDim.x + threadIdx.x;
    if (i >= total) return;
    int c = i & 255;
    float v = fmaxf(agg[i] + bias[c], 0.f);
    out[i] = v;
    if (outbf) outbf[i] = f2bf(v);
}

// ---------------------------------------------------------------------------
// Pooling
// ---------------------------------------------------------------------------
__global__ void k_pool_accum(const float* __restrict__ x,
                             const float* __restrict__ h1,
                             const float* __restrict__ h2,
                             const int* __restrict__ batch,
                             float* __restrict__ pooled,   // [G,640] init 0
                             int total) {                  // N*640
    int i = blockIdx.x * blockDim.x + threadIdx.x;
    if (i >= total) return;
    int n = i / DCAT, c = i - n * DCAT;
    int g = batch[n];
    float v;
    if      (c < IN_CH)        v = x [(size_t)n * IN_CH + c];
    else if (c < IN_CH + HD)   v = h1[(size_t)n * HD + (c - IN_CH)];
    else                       v = h2[(size_t)n * HD + (c - IN_CH - HD)];
    atomicAdd(&pooled[g * DCAT + c], v);
}

__global__ void k_pool_count(const int* __restrict__ batch,
                             float* __restrict__ cnt, int n) {
    int i = blockIdx.x * blockDim.x + threadIdx.x;
    if (i < n) atomicAdd(&cnt[batch[i]], 1.f);
}

__global__ void k_pool_div(float* __restrict__ pooled,
                           const float* __restrict__ cnt,
                           unsigned short* __restrict__ pooledbf, int total) {
    int i = blockIdx.x * blockDim.x + threadIdx.x;
    if (i >= total) return;
    int g = i / DCAT;
    float v = pooled[i] / fmaxf(cnt[g], 1.f);
    pooled[i]   = v;
    pooledbf[i] = f2bf(v);
}

// ---------------------------------------------------------------------------
// Host-side orchestration
// ---------------------------------------------------------------------------
static inline unsigned int cdiv(long long a, long long b) {
    return (unsigned int)((a + b - 1) / b);
}

extern "C" void kernel_launch(void* const* d_in, const int* in_sizes, int n_in,
                              void* d_out, int out_size, void* d_ws, size_t ws_size,
                              hipStream_t stream) {
    (void)in_sizes; (void)n_in; (void)out_size; (void)ws_size;
    const float* x    = (const float*)d_in[0];
    const int*   ei   = (const int*)  d_in[1];   // [2,E]
    const int*   bat  = (const int*)  d_in[2];   // [N]
    const float* W1   = (const float*)d_in[3];
    const float* a1s  = (const float*)d_in[4];
    const float* a1d  = (const float*)d_in[5];
    const float* b1   = (const float*)d_in[6];
    const float* W2   = (const float*)d_in[7];
    const float* a2s  = (const float*)d_in[8];
    const float* a2d  = (const float*)d_in[9];
    const float* b2   = (const float*)d_in[10];
    const float* W3   = (const float*)d_in[11];
    const float* b3   = (const float*)d_in[12];
    const float* W4   = (const float*)d_in[13];
    const float* b4   = (const float*)d_in[14];
    float* out = (float*)d_out;                  // [G,128]

    // ---- carve workspace ----
    char* wp = (char*)d_ws;
    auto alloc = [&](size_t bytes) -> char* {
        char* p = wp;
        wp += (bytes + 255) & ~(size_t)255;
        return p;
    };
    unsigned short* xb     = (unsigned short*)alloc((size_t)NN * IN_CH * 2);
    unsigned short* Wb     = (unsigned short*)alloc((size_t)DCAT * HD * 2);   // max weight (transposed)
    float*          hpre   = (float*)alloc((size_t)NN * HD * 4);
    float*          agg    = (float*)alloc((size_t)NN * HD * 4);
    float*          h1     = (float*)alloc((size_t)NN * HD * 4);
    unsigned short* h1b    = (unsigned short*)alloc((size_t)NN * HD * 2);
    float*          h2     = (float*)alloc((size_t)NN * HD * 4);
    float*          esrc   = (float*)alloc((size_t)NN * HEADS * 4);
    float*          edst   = (float*)alloc((size_t)NN * HEADS * 4);
    float*          mbuf   = (float*)alloc((size_t)NN * HEADS * 4);
    float*          dnm    = (float*)alloc((size_t)NN * HEADS * 4);
    float*          ebuf   = (float*)alloc((size_t)EE * HEADS * 4);
    float*          pooled = (float*)alloc((size_t)GG * DCAT * 4);
    unsigned short* pooledb= (unsigned short*)alloc((size_t)GG * DCAT * 2);
    float*          cnt    = (float*)alloc((size_t)GG * 4);
    float*          hmlp   = (float*)alloc((size_t)GG * 256 * 4);
    unsigned short* hmlpb  = (unsigned short*)alloc((size_t)GG * 256 * 2);

    const int BT  = 256;
    const int nh  = NN * HEADS;
    const int eh  = EE * HEADS;
    const int nhd = NN * HD;

    // ======================= Layer 1 =======================
    k_f32_to_bf16<<<cdiv((long long)NN * IN_CH, BT), BT, 0, stream>>>(x, xb, NN * IN_CH);
    k_transpose_bf16<<<cdiv(IN_CH * HD, BT), BT, 0, stream>>>(W1, Wb, IN_CH, HD);
    {
        dim3 g(cdiv(NN, 64), HD / 128);
        k_gemm_bf16_wmma<<<g, BT, 0, stream>>>(xb, Wb, hpre, nullptr, nullptr,
                                               NN, HD, IN_CH, 0);
    }
    k_node_scores<<<cdiv(nh, BT), BT, 0, stream>>>(hpre, a1s, a1d, esrc, edst, NN);
    k_fill_f32<<<cdiv(nh, BT), BT, 0, stream>>>(mbuf, -INFINITY, nh);
    k_fill_f32<<<cdiv(nh, BT), BT, 0, stream>>>(dnm, 0.f, nh);
    k_edge_lrelu_max<<<cdiv(eh, BT), BT, 0, stream>>>(ei, esrc, edst, ebuf, mbuf, eh);
    k_fix_m<<<cdiv(nh, BT), BT, 0, stream>>>(mbuf, nh);
    k_edge_exp_sum<<<cdiv(eh, BT), BT, 0, stream>>>(ei, mbuf, ebuf, dnm, eh);
    k_fill_f32<<<cdiv(nhd, BT), BT, 0, stream>>>(agg, 0.f, nhd);
    k_edge_scatter<<<EE, BT, 0, stream>>>(ei, hpre, ebuf, dnm, agg);
    k_bias_relu<<<cdiv(nhd, BT), BT, 0, stream>>>(agg, b1, h1, h1b, nhd);

    // ======================= Layer 2 =======================
    k_transpose_bf16<<<cdiv(HD * HD, BT), BT, 0, stream>>>(W2, Wb, HD, HD);
    {
        dim3 g(cdiv(NN, 64), HD / 128);
        k_gemm_bf16_wmma<<<g, BT, 0, stream>>>(h1b, Wb, hpre, nullptr, nullptr,
                                               NN, HD, HD, 0);
    }
    k_node_scores<<<cdiv(nh, BT), BT, 0, stream>>>(hpre, a2s, a2d, esrc, edst, NN);
    k_fill_f32<<<cdiv(nh, BT), BT, 0, stream>>>(mbuf, -INFINITY, nh);
    k_fill_f32<<<cdiv(nh, BT), BT, 0, stream>>>(dnm, 0.f, nh);
    k_edge_lrelu_max<<<cdiv(eh, BT), BT, 0, stream>>>(ei, esrc, edst, ebuf, mbuf, eh);
    k_fix_m<<<cdiv(nh, BT), BT, 0, stream>>>(mbuf, nh);
    k_edge_exp_sum<<<cdiv(eh, BT), BT, 0, stream>>>(ei, mbuf, ebuf, dnm, eh);
    k_fill_f32<<<cdiv(nhd, BT), BT, 0, stream>>>(agg, 0.f, nhd);
    k_edge_scatter<<<EE, BT, 0, stream>>>(ei, hpre, ebuf, dnm, agg);
    k_bias_relu<<<cdiv(nhd, BT), BT, 0, stream>>>(agg, b2, h2, nullptr, nhd);

    // ======================= Pooling =======================
    k_fill_f32<<<cdiv(GG * DCAT, BT), BT, 0, stream>>>(pooled, 0.f, GG * DCAT);
    k_fill_f32<<<1, GG, 0, stream>>>(cnt, 0.f, GG);
    k_pool_accum<<<cdiv((long long)NN * DCAT, BT), BT, 0, stream>>>(x, h1, h2, bat,
                                                                    pooled, NN * DCAT);
    k_pool_count<<<cdiv(NN, BT), BT, 0, stream>>>(bat, cnt, NN);
    k_pool_div<<<cdiv(GG * DCAT, BT), BT, 0, stream>>>(pooled, cnt, pooledb, GG * DCAT);

    // ======================= MLP =======================
    k_transpose_bf16<<<cdiv(DCAT * 256, BT), BT, 0, stream>>>(W3, Wb, DCAT, 256);
    {
        dim3 g(cdiv(GG, 64), 256 / 128);
        k_gemm_bf16_wmma<<<g, BT, 0, stream>>>(pooledb, Wb, hmlp, hmlpb, b3,
                                               GG, 256, DCAT, 1);
    }
    k_transpose_bf16<<<cdiv(256 * 128, BT), BT, 0, stream>>>(W4, Wb, 256, 128);
    {
        dim3 g(cdiv(GG, 64), 128 / 128);
        k_gemm_bf16_wmma<<<g, BT, 0, stream>>>(hmlpb, Wb, out, nullptr, b4,
                                               GG, 128, 256, 0);
    }
}
